// OptionEncoder_Recurrent_35725537968414
// MI455X (gfx1250) — matrix operand order repaired and checked by reading, not compiled, not run on hardware
//
#include <hip/hip_runtime.h>
#include <hip/hip_bf16.h>
#include <math.h>

// Problem constants (from reference)
#define BB   256
#define TT   1024
#define INP  64
#define HH   32
#define LL   2
#define CC   4
#define LC   8          // L*C
#define IN0  72         // INP + LC
// 1/tau = 10; log2(e) = 1.4426950408889634
#define LOG2E      1.4426950408889634f
#define TAUINV_L2E 14.426950408889634f   // (1/tau)*log2(e)

// Tile / LDS geometry
#define MT   16         // batch rows per workgroup (WMMA M)
#define AS   136        // Abuf row stride (bf16), 272B = 16B-aligned rows
#define A1S  72         // A1buf row stride (bf16), 144B = 16B-aligned rows
#define ZS   132        // zbuf row stride (f32)

typedef __attribute__((ext_vector_type(16))) __bf16 v16bf;
typedef __attribute__((ext_vector_type(8)))  __bf16 v8bf;
typedef __attribute__((ext_vector_type(8)))  float  v8f;

// Branch-free fast math on native v_exp_f32 (2^x) / v_rcp_f32.
__device__ __forceinline__ float fast_sig(float x) {
    // 1/(1+e^-x) = 1/(1+2^(-x*log2e));  exp2 saturates cleanly at both ends.
    return __builtin_amdgcn_rcpf(1.0f + __builtin_amdgcn_exp2f(-LOG2E * x));
}
__device__ __forceinline__ float fast_tanh(float x) {
    // tanh(x) = 1 - 2/(e^(2x)+1) = 1 - 2/(2^(2x*log2e)+1)
    return fmaf(-2.0f,
                __builtin_amdgcn_rcpf(1.0f + __builtin_amdgcn_exp2f(2.0f * LOG2E * x)),
                1.0f);
}

// Load one 16x32 bf16 A-tile from LDS in contiguous k order (same order as the B tiles):
// lanes 0-15: elems 0..15 = K[0..15]
// lanes 16-31: elems 0..15 = K[16..31]
__device__ __forceinline__ v16bf load_a_tile(const __bf16* row0, int stride,
                                             int kbase, int lane) {
    const __bf16* p = row0 + (lane & 15) * stride + kbase + ((lane >> 4) << 4);
    v8bf lo = *(const v8bf*)p;
    v8bf hi = *(const v8bf*)(p + 8);
    return __builtin_shufflevector(lo, hi, 0,1,2,3,4,5,6,7,8,9,10,11,12,13,14,15);
}

__global__ __launch_bounds__(256)
void option_encoder_recurrent_kernel(
    const float* __restrict__ states,   // [B,T,IN]
    const float* __restrict__ gumbel,   // [T,B,L,C]
    const float* __restrict__ W_ih0,    // [128,72]
    const float* __restrict__ W_hh0,    // [128,32]
    const float* __restrict__ b_ih0,    // [128]
    const float* __restrict__ b_hh0,    // [128]
    const float* __restrict__ W_ih1,    // [128,32]
    const float* __restrict__ W_hh1,    // [128,32]
    const float* __restrict__ b_ih1,    // [128]
    const float* __restrict__ b_hh1,    // [128]
    const float* __restrict__ W_lin,    // [8,32]
    const float* __restrict__ b_lin,    // [8]
    float* __restrict__ out)            // [B,T,8]
{
    __shared__ __align__(16) __bf16 Abuf[MT * AS];    // [x(64) | ct(8) | h0(32) | 0(24)]
    __shared__ __align__(16) __bf16 A1buf[MT * A1S];  // [h0(32) | h1(32) | pad]
    __shared__ float zbuf[MT * ZS];                   // gate pre-activations 16x128
    __shared__ float c0buf[MT * HH];
    __shared__ float c1buf[MT * HH];
    __shared__ float ctbuf[MT * LC];
    __shared__ float qbuf[MT * 16];
    __shared__ float bias0[128];
    __shared__ float bias1[128];
    __shared__ float blin[LC];

    const int tid  = threadIdx.x;
    const int wv   = tid >> 5;         // wave id 0..7  (owns gate cols 16*wv..16*wv+15)
    const int lane = tid & 31;
    const int m0   = blockIdx.x * MT;  // first batch row of this workgroup

    // ---------------- init LDS ----------------
    for (int i = tid; i < MT * AS; i += 256)  Abuf[i]  = (__bf16)0.0f;
    for (int i = tid; i < MT * A1S; i += 256) A1buf[i] = (__bf16)0.0f;
    for (int i = tid; i < MT * HH; i += 256) { c0buf[i] = 0.0f; c1buf[i] = 0.0f; }
    for (int i = tid; i < MT * LC; i += 256)  ctbuf[i] = ((i & 3) == 0) ? 1.0f : 0.0f; // one-hot cat 0
    if (tid < 128) { bias0[tid] = b_ih0[tid] + b_hh0[tid];
                     bias1[tid] = b_ih1[tid] + b_hh1[tid]; }
    if (tid < LC)  blin[tid] = b_lin[tid];

    // ---------------- weights -> registers (contiguous k order, same as load_a_tile) ----------------
    // B (KxN=32x16 per tile): lane n (0-15) holds K=0..15 of column n,
    // lane n+16 holds K=16..31 of column n.
    const int ncol = (wv << 4) + (lane & 15);   // gate/output column 0..127
    const int khi  = (lane >> 4) << 4;          // 0 or 16

    v16bf B0[4];   // layer0: K = [x(0..63)|ct(64..71)|h(72..103)|0], 4 k-tiles
#pragma unroll
    for (int kt = 0; kt < 4; ++kt) {
#pragma unroll
        for (int e = 0; e < 16; ++e) {
            int k = kt * 32 + khi + e;
            float w = (k < IN0) ? W_ih0[ncol * IN0 + k]
                    : (k < IN0 + HH) ? W_hh0[ncol * HH + (k - IN0)]
                    : 0.0f;
            B0[kt][e] = (__bf16)w;
        }
    }
    v16bf B1[2];   // layer1: K = [h0(0..31)|h1(32..63)], 2 k-tiles
#pragma unroll
    for (int kt = 0; kt < 2; ++kt) {
#pragma unroll
        for (int e = 0; e < 16; ++e) {
            int k = kt * 32 + khi + e;
            float w = (k < HH) ? W_ih1[ncol * HH + k]
                               : W_hh1[ncol * HH + (k - HH)];
            B1[kt][e] = (__bf16)w;
        }
    }
    v16bf BLin;    // head: K=32, N: cols 0..7 valid, 8..15 zero
    {
        int n8 = lane & 15;
#pragma unroll
        for (int e = 0; e < 16; ++e) {
            int k = khi + e;
            BLin[e] = (__bf16)((n8 < LC) ? W_lin[n8 * HH + k] : 0.0f);
        }
    }
    __syncthreads();

    // ---------------- time loop (serial recurrence) ----------------
    for (int t = 0; t < TT; ++t) {
        int tn = (t + 1 < TT) ? (t + 1) : t;   // prefetch target (clamped)

        // Phase A: build layer-0 A = [x_t | ct | h0_prev]  (h0 region written last step)
        // + prefetch next step's x into cache (global_prefetch_b8).
#pragma unroll
        for (int r = 0; r < 4; ++r) {               // x: 16x64 = 1024 elems
            int e = r * 256 + tid;
            int m = e >> 6, k = e & 63;
            const float* src = states + ((size_t)(m0 + m) * TT + t) * INP + k;
            Abuf[m * AS + k] = (__bf16)(*src);
            __builtin_prefetch(states + ((size_t)(m0 + m) * TT + tn) * INP + k, 0, 3);
        }
        if (tid < MT * LC) {                        // ct: 16x8
            int m = tid >> 3, j = tid & 7;
            Abuf[m * AS + INP + j] = (__bf16)ctbuf[m * LC + j];
            __builtin_prefetch(gumbel + ((size_t)(tn * BB + (m0 + m)) * LL) * CC + j, 0, 3);
        }
        __syncthreads();

        // Phase B: z0 = A(16x128) x B0(128x128) per-wave N-slice
        {
            v8f acc = {};
#pragma unroll
            for (int kt = 0; kt < 4; ++kt) {
                v16bf a = load_a_tile(Abuf, AS, kt * 32, lane);
                acc = __builtin_amdgcn_wmma_f32_16x16x32_bf16(
                    false, a, false, B0[kt], (short)0, acc, false, false);
            }
            int rbase = (lane >> 4) << 3;           // D layout: lanes 0-15 M0..7, 16-31 M8..15
#pragma unroll
            for (int r = 0; r < 8; ++r)
                zbuf[(rbase + r) * ZS + ncol] = acc[r];
        }
        __syncthreads();

        // Phase C: layer-0 gates (i,f,g,o), update c0 / h0
#pragma unroll
        for (int r = 0; r < 2; ++r) {               // 16x32 = 512 elems
            int e = r * 256 + tid;
            int m = e >> 5, j = e & 31;
            float zi = zbuf[m * ZS + j]      + bias0[j];
            float zf = zbuf[m * ZS + 32 + j] + bias0[32 + j];
            float zg = zbuf[m * ZS + 64 + j] + bias0[64 + j];
            float zo = zbuf[m * ZS + 96 + j] + bias0[96 + j];
            float c  = fast_sig(zf) * c0buf[m * HH + j] + fast_sig(zi) * fast_tanh(zg);
            float h  = fast_sig(zo) * fast_tanh(c);
            c0buf[m * HH + j] = c;
            __bf16 hb = (__bf16)h;
            A1buf[m * A1S + j]      = hb;   // layer-1 input (this step)
            Abuf [m * AS + IN0 + j] = hb;   // layer-0 recurrent input (next step)
        }
        __syncthreads();

        // Phase D: z1 = [h0|h1_prev](16x64) x B1(64x128)
        {
            v8f acc = {};
#pragma unroll
            for (int kt = 0; kt < 2; ++kt) {
                v16bf a = load_a_tile(A1buf, A1S, kt * 32, lane);
                acc = __builtin_amdgcn_wmma_f32_16x16x32_bf16(
                    false, a, false, B1[kt], (short)0, acc, false, false);
            }
            int rbase = (lane >> 4) << 3;
#pragma unroll
            for (int r = 0; r < 8; ++r)
                zbuf[(rbase + r) * ZS + ncol] = acc[r];
        }
        __syncthreads();

        // Phase E: layer-1 gates, update c1 / h1
#pragma unroll
        for (int r = 0; r < 2; ++r) {
            int e = r * 256 + tid;
            int m = e >> 5, j = e & 31;
            float zi = zbuf[m * ZS + j]      + bias1[j];
            float zf = zbuf[m * ZS + 32 + j] + bias1[32 + j];
            float zg = zbuf[m * ZS + 64 + j] + bias1[64 + j];
            float zo = zbuf[m * ZS + 96 + j] + bias1[96 + j];
            float c  = fast_sig(zf) * c1buf[m * HH + j] + fast_sig(zi) * fast_tanh(zg);
            float h  = fast_sig(zo) * fast_tanh(c);
            c1buf[m * HH + j] = c;
            A1buf[m * A1S + 32 + j] = (__bf16)h;  // for q now + recurrence next step
        }
        __syncthreads();

        // Phase F: q = h1(16x32) x W_lin^T (N padded to 16), wave 0 only
        if (wv == 0) {
            v16bf a = load_a_tile(A1buf, A1S, 32, lane);
            v8f q = {};
            q = __builtin_amdgcn_wmma_f32_16x16x32_bf16(
                false, a, false, BLin, (short)0, q, false, false);
            int rbase = (lane >> 4) << 3;
            int n8 = lane & 15;
#pragma unroll
            for (int r = 0; r < 8; ++r)
                qbuf[(rbase + r) * 16 + n8] = q[r];
        }
        __syncthreads();

        // Phase G: Gumbel-softmax over C=4 per (row, l); write ct and output
        if (tid < MT * LL) {                        // 32 threads: one (m,l) group each
            int m = tid >> 1, l = tid & 1;
            const float* g = gumbel + ((size_t)(t * BB + (m0 + m)) * LL + l) * CC;
            float v[CC];
            float mx = -1e30f;
#pragma unroll
            for (int c = 0; c < CC; ++c) {
                // logits/tau, directly in log2 domain: exp(x/tau) = 2^(x * (1/tau)*log2e)
                v[c] = (qbuf[m * 16 + l * CC + c] + blin[l * CC + c] + g[c]) * TAUINV_L2E;
                mx = fmaxf(mx, v[c]);
            }
            float s = 0.0f;
#pragma unroll
            for (int c = 0; c < CC; ++c) {
                v[c] = __builtin_amdgcn_exp2f(v[c] - mx);
                s += v[c];
            }
            float inv = __builtin_amdgcn_rcpf(s);
            float* o = out + ((size_t)(m0 + m) * TT + t) * LC + l * CC;
#pragma unroll
            for (int c = 0; c < CC; ++c) {
                float y = v[c] * inv;
                ctbuf[m * LC + l * CC + c] = y;
                o[c] = y;
            }
        }
        __syncthreads();
    }
}

extern "C" void kernel_launch(void* const* d_in, const int* in_sizes, int n_in,
                              void* d_out, int out_size, void* d_ws, size_t ws_size,
                              hipStream_t stream) {
    (void)in_sizes; (void)n_in; (void)out_size; (void)d_ws; (void)ws_size;
    const float* states = (const float*)d_in[0];
    const float* gumbel = (const float*)d_in[1];
    const float* W_ih0  = (const float*)d_in[2];
    const float* W_hh0  = (const float*)d_in[3];
    const float* b_ih0  = (const float*)d_in[4];
    const float* b_hh0  = (const float*)d_in[5];
    const float* W_ih1  = (const float*)d_in[6];
    const float* W_hh1  = (const float*)d_in[7];
    const float* b_ih1  = (const float*)d_in[8];
    const float* b_hh1  = (const float*)d_in[9];
    const float* W_lin  = (const float*)d_in[10];
    const float* b_lin  = (const float*)d_in[11];
    float* out = (float*)d_out;

    dim3 grid(BB / MT);   // 16 workgroups, one 16-row batch tile each
    dim3 block(256);      // 8 waves: one 16-col gate slice per wave
    option_encoder_recurrent_kernel<<<grid, block, 0, stream>>>(
        states, gumbel, W_ih0, W_hh0, b_ih0, b_hh0,
        W_ih1, W_hh1, b_ih1, b_hh1, W_lin, b_lin, out);
}
